// ROIHead_66065186947204
// MI455X (gfx1250) — matrix operand order repaired and checked by reading, not compiled, not run on hardware
//
#include <hip/hip_runtime.h>
#include <hip/hip_bf16.h>
#include <math.h>

// ---------------------------------------------------------------------------
// Types for CDNA5 WMMA / async DMA
// ---------------------------------------------------------------------------
typedef __attribute__((ext_vector_type(16))) __bf16 v16bf;
typedef __attribute__((ext_vector_type(8)))  float  v8f;
typedef int v4i_t __attribute__((vector_size(16)));   // matches builtin proto
typedef __attribute__((address_space(3))) v4i_t* lds_v4i_p;
typedef __attribute__((address_space(1))) v4i_t* glob_v4i_p;

// Problem constants (fixed by setup_inputs)
#define NROI   2000
#define MPAD   2048
#define CCH    256
#define FH     200
#define FW     200
#define PP     7
#define PSZ    64
#define DDIM   12544   // 256*49
#define FDIM   1024
#define NCLS   21
#define NFG    20
#define NCAND  40000   // NROI*NFG
#define NPAD   65536
#define PRENMS 2000
#define TOPK   100
#define BBOX_CLIP 4.135166556742356f // log(1000/16)

// ---------------------------------------------------------------------------
// CDNA5 async global->LDS copy (ASYNCcnt-tracked) with inline-asm fallback
// ---------------------------------------------------------------------------
__device__ __forceinline__ void async_copy_b128(lds_v4i_p lds, const void* glob)
{
#if __has_builtin(__builtin_amdgcn_global_load_async_to_lds_b128)
    __builtin_amdgcn_global_load_async_to_lds_b128(
        (glob_v4i_p)(v4i_t*)(void*)glob, lds, 0, 0);
#else
    unsigned loff = (unsigned)(uintptr_t)lds;
    unsigned long long g = (unsigned long long)(uintptr_t)glob;
    asm volatile("global_load_async_to_lds_b128 %0, %1, off"
                 :: "v"(loff), "v"(g) : "memory");
#endif
}

template <int N>
__device__ __forceinline__ void wait_asynccnt()
{
#if __has_builtin(__builtin_amdgcn_s_wait_asynccnt)
    __builtin_amdgcn_s_wait_asynccnt(N);
#else
    asm volatile("s_wait_asynccnt %0" :: "n"(N) : "memory");
#endif
}

// ---------------------------------------------------------------------------
// 0) Weight conversion fp32 [N,K] row-major  ->  bf16 fragment-packed B.
//    Packed layout: block b = nt*(K/32)+kc is 512 halfs (1 KB).
//    Within block, half (L,i) holds  B[k = kc*32 + i + 16*(L/16)][n = nt*16 + L%16]
//    = W[n][k].  Contiguous k order, same as the GEMM's A fragments, so the GEMM loads
//    one v16bf per lane per fragment (2x global_load_b128, L2-resident).
// ---------------------------------------------------------------------------
__global__ void __launch_bounds__(256)
k_pack_b(const float* __restrict__ W, __bf16* __restrict__ Bp, int N_, int K_)
{
    long t = (long)blockIdx.x * blockDim.x + threadIdx.x;
    long total = (long)N_ * K_;
    if (t >= total) return;
    int  KC = K_ >> 5;                 // k-chunks of 32
    long b = t >> 9;                   // 512 halfs per block
    int  r = (int)(t & 511);
    int  L = r >> 4;                   // lane 0..31
    int  i = r & 15;                   // half index 0..15
    long nt = b / KC;
    long kc = b % KC;
    int  k = (int)(kc * 32) + i + 16 * (L >> 4);
    int  n = (int)(nt * 16) + (L & 15);
    Bp[t] = (__bf16)W[(size_t)n * K_ + k];
}

// zero the padded rows of pooled so the GEMM needs no row guard
__global__ void __launch_bounds__(256)
k_zero_pad(__bf16* __restrict__ p)
{
    long i = (long)blockIdx.x * blockDim.x + threadIdx.x;
    long total = (long)(MPAD - NROI) * DDIM;
    if (i < total) p[(size_t)NROI * DDIM + i] = (__bf16)0.f;
}

// ---------------------------------------------------------------------------
// 1) ROI max-pool (torchvision roi_pool quantization, 64x64 patch clamp)
//    one block per ROI, one thread per channel -> pooled bf16 [MPAD, DDIM]
// ---------------------------------------------------------------------------
__global__ void __launch_bounds__(256)
k_roi_pool(const float* __restrict__ feat, const float* __restrict__ rois,
           const int* __restrict__ imw_p, __bf16* __restrict__ pooled)
{
    int n = blockIdx.x;
    int c = threadIdx.x;
    float scale = (float)imw_p[0] / (float)FW;
    float x1 = roundf(rois[n * 4 + 0] * scale);
    float y1 = roundf(rois[n * 4 + 1] * scale);
    float x2 = roundf(rois[n * 4 + 2] * scale);
    float y2 = roundf(rois[n * 4 + 3] * scale);
    float bw = fmaxf(x2 - x1 + 1.f, 1.f) / (float)PP;
    float bh = fmaxf(y2 - y1 + 1.f, 1.f) / (float)PP;
    int sh = min(max((int)y1, 0), FH - PSZ);
    int sw = min(max((int)x1, 0), FW - PSZ);
    const float* fc = feat + (size_t)c * FH * FW;
    for (int ph = 0; ph < PP; ++ph) {
        int hs = (int)fminf(fmaxf(floorf((float)ph * bh) + y1, 0.f), (float)FH);
        int he = (int)fminf(fmaxf(ceilf((float)(ph + 1) * bh) + y1, 0.f), (float)FH);
        int hlo = max(hs, sh), hhi = min(he, sh + PSZ);
        for (int pw = 0; pw < PP; ++pw) {
            int ws_ = (int)fminf(fmaxf(floorf((float)pw * bw) + x1, 0.f), (float)FW);
            int we_ = (int)fminf(fmaxf(ceilf((float)(pw + 1) * bw) + x1, 0.f), (float)FW);
            int wlo = max(ws_, sw), whi = min(we_, sw + PSZ);
            float m = -1e30f;
            for (int h = hlo; h < hhi; ++h) {
                const float* rowp = fc + (size_t)h * FW;
                for (int w = wlo; w < whi; ++w) m = fmaxf(m, rowp[w]);
            }
            if (m <= -5e29f) m = 0.f;   // empty bin -> 0
            pooled[(size_t)n * DDIM + c * 49 + ph * PP + pw] = (__bf16)m;
        }
    }
}

// ---------------------------------------------------------------------------
// 2) bf16 WMMA GEMM:  out[M,N] = relu(A[M,K] * B[K,N] + bias)
//    A: double-buffered LDS tiles filled with async global->LDS DMA
//       (GLOBAL_LOAD_ASYNC_TO_LDS_B128, ASYNCcnt), overlapped with compute.
//    B: read directly from the fragment-packed global layout (L2-resident).
//    Block tile 128x128, 8 waves (2x4), each wave 64x32 via 4x2 WMMA tiles.
//    Last k-chunk peeled so the steady-state loop is branch-free.
// ---------------------------------------------------------------------------
#define BM 128
#define BN 128
#define KB 32

template <bool OUT_BF16>
__global__ void __launch_bounds__(256)
k_gemm_bf16(const __bf16* __restrict__ A, const __bf16* __restrict__ Bp,
            const float* __restrict__ bias, void* __restrict__ outp,
            int M, int N, int K)
{
    __shared__ __attribute__((aligned(16))) __bf16 As[2][BM][KB + 8];

    const int tid  = threadIdx.x;
    const int lane = tid & 31;
    const int wave = tid >> 5;     // 0..7
    const int wm   = wave >> 2;    // 0..1 (M)
    const int wn   = wave & 3;     // 0..3 (N)
    const int m0   = blockIdx.y * BM;
    const int n0   = blockIdx.x * BN;
    const int g    = lane >> 4;    // lane group
    const int lr   = lane & 15;
    const int KC   = K >> 5;       // k-chunks

    v8f acc[4][2];
    for (int a = 0; a < 4; ++a)
        for (int b = 0; b < 2; ++b)
            for (int e = 0; e < 8; ++e) acc[a][b][e] = 0.f;

    // per-wave packed-B base pointers (v16bf = 32B per lane per fragment)
    const v16bf* bbase[2];
    #pragma unroll
    for (int nt = 0; nt < 2; ++nt) {
        size_t ntile = (size_t)(n0 >> 4) + wn * 2 + nt;
        bbase[nt] = (const v16bf*)(Bp + (ntile * KC) * 512) + lane;
    }

    // per-thread A staging addresses: 16 contiguous bf16 (32 B) per thread.
    // LDS destinations pre-converted to addrspace(3) so the hot loop has no
    // generic->LDS casts (no per-iteration null-check cndmasks).
    const int se = tid * 16;
    const int sr = se >> 5, sc = se & 31;
    const __bf16* asrc = A + (size_t)(m0 + sr) * K + sc;
    lds_v4i_p adst[2] = { (lds_v4i_p)(v4i_t*)(void*)&As[0][sr][sc],
                          (lds_v4i_p)(v4i_t*)(void*)&As[1][sr][sc] };

    // per-k-chunk compute on buffer `cur`
    auto compute = [&](int kc, int cur) {
        // B fragments: direct packed global loads
        v16bf bfrag[2];
        #pragma unroll
        for (int nt = 0; nt < 2; ++nt) bfrag[nt] = bbase[nt][(size_t)kc * 32];
        // A fragments (M = lane%16, K = i + 16g, same contiguous k order as packed B)
        #pragma unroll
        for (int mt = 0; mt < 4; ++mt) {
            v16bf afrag;
            int arow = wm * 64 + mt * 16 + lr;
            #pragma unroll
            for (int i = 0; i < 16; ++i) {
                int kk = i + 16 * g;
                afrag[i] = As[cur][arow][kk];
            }
            #pragma unroll
            for (int nt = 0; nt < 2; ++nt) {
                acc[mt][nt] = __builtin_amdgcn_wmma_f32_16x16x32_bf16(
                    false, afrag, false, bfrag[nt], (short)0, acc[mt][nt],
                    false, false);
            }
        }
    };

    // prologue: stage k-chunk 0 into buffer 0
    async_copy_b128(adst[0],     asrc);
    async_copy_b128(adst[0] + 1, asrc + 8);

    // steady state: branch-free stage-ahead + compute
    for (int kc = 0; kc + 1 < KC; ++kc) {
        const int cur = kc & 1;
        const __bf16* s = asrc + (size_t)(kc + 1) * KB;
        async_copy_b128(adst[cur ^ 1],     s);
        async_copy_b128(adst[cur ^ 1] + 1, s + 8);
        wait_asynccnt<2>();   // in-order: oldest 2 (current tile) done
        __syncthreads();
        compute(kc, cur);
        __syncthreads();      // readers done before this buffer is re-staged
    }
    // peeled final chunk
    wait_asynccnt<0>();
    __syncthreads();
    compute(KC - 1, (KC - 1) & 1);

    // Epilogue: C layout — N = lane%16, M = r + 8*(lane/16)
    __bf16* outh = (__bf16*)outp;
    float*  outf = (float*)outp;
    #pragma unroll
    for (int mt = 0; mt < 4; ++mt)
        #pragma unroll
        for (int nt = 0; nt < 2; ++nt)
            #pragma unroll
            for (int r = 0; r < 8; ++r) {
                int row = m0 + wm * 64 + mt * 16 + r + 8 * g;
                int col = n0 + wn * 32 + nt * 16 + lr;
                float v = fmaxf(acc[mt][nt][r] + bias[col], 0.f);
                if (OUT_BF16) outh[(size_t)row * N + col] = (__bf16)v;
                else          outf[(size_t)row * N + col] = v;
            }
}

// ---------------------------------------------------------------------------
// 3) cls/reg heads: heads[n,0..20]=logits, heads[n,21..104]=reg (fp32 dot)
// ---------------------------------------------------------------------------
__global__ void __launch_bounds__(128)
k_heads(const float* __restrict__ X, const float* __restrict__ cls_w,
        const float* __restrict__ cls_b, const float* __restrict__ reg_w,
        const float* __restrict__ reg_b, float* __restrict__ heads)
{
    __shared__ float xs[FDIM];
    int n = blockIdx.x;
    for (int i = threadIdx.x; i < FDIM; i += 128) xs[i] = X[(size_t)n * FDIM + i];
    __syncthreads();
    int c = threadIdx.x;
    if (c < 105) {
        const float* wrow;
        float b;
        if (c < NCLS) { wrow = cls_w + (size_t)c * FDIM;          b = cls_b[c]; }
        else          { wrow = reg_w + (size_t)(c - NCLS) * FDIM; b = reg_b[c - NCLS]; }
        float s = b;
        for (int k = 0; k < FDIM; ++k) s += xs[k] * wrow[k];
        heads[(size_t)n * 105 + c] = s;
    }
}

// ---------------------------------------------------------------------------
// 4) softmax + box decode + score/min-size filter -> candidate arrays
// ---------------------------------------------------------------------------
__global__ void __launch_bounds__(256)
k_decode(const float* __restrict__ heads, const float* __restrict__ rois,
         const int* __restrict__ ih_p, const int* __restrict__ iw_p,
         float* __restrict__ cscore, float* __restrict__ cbox)
{
    int n = blockIdx.x * blockDim.x + threadIdx.x;
    if (n >= NROI) return;
    const float* hh = heads + (size_t)n * 105;
    float mx = hh[0];
    for (int j = 1; j < NCLS; ++j) mx = fmaxf(mx, hh[j]);
    float ex[NCLS], den = 0.f;
    for (int j = 0; j < NCLS; ++j) { ex[j] = expf(hh[j] - mx); den += ex[j]; }
    float imw = (float)iw_p[0], imh = (float)ih_p[0];
    float px1 = rois[n * 4 + 0], py1 = rois[n * 4 + 1];
    float px2 = rois[n * 4 + 2], py2 = rois[n * 4 + 3];
    float pw = px2 - px1, phh = py2 - py1;
    float pcx = px1 + 0.5f * pw, pcy = py1 + 0.5f * phh;
    for (int j = 0; j < NFG; ++j) {
        int cls = j + 1;
        float dx = hh[21 + cls * 4 + 0];
        float dy = hh[21 + cls * 4 + 1];
        float dw = fminf(hh[21 + cls * 4 + 2], BBOX_CLIP);
        float dh = fminf(hh[21 + cls * 4 + 3], BBOX_CLIP);
        float cx = dx * pw + pcx, cy = dy * phh + pcy;
        float w = expf(dw) * pw, h = expf(dh) * phh;
        float bx1 = fminf(fmaxf(cx - 0.5f * w, 0.f), imw);
        float by1 = fminf(fmaxf(cy - 0.5f * h, 0.f), imh);
        float bx2 = fminf(fmaxf(cx + 0.5f * w, 0.f), imw);
        float by2 = fminf(fmaxf(cy + 0.5f * h, 0.f), imh);
        float sc = ex[cls] / den;
        bool valid = (sc > 0.05f) && ((bx2 - bx1) >= 16.f) && ((by2 - by1) >= 16.f);
        int idx = n * NFG + j;
        cscore[idx] = valid ? sc : -1.f;
        cbox[idx * 4 + 0] = bx1; cbox[idx * 4 + 1] = by1;
        cbox[idx * 4 + 2] = bx2; cbox[idx * 4 + 3] = by2;
    }
}

// ---------------------------------------------------------------------------
// 5) sort init + global bitonic sort (desc by score, asc by index on ties)
// ---------------------------------------------------------------------------
__global__ void __launch_bounds__(256)
k_sort_init(float* __restrict__ key, int* __restrict__ idx)
{
    int i = blockIdx.x * blockDim.x + threadIdx.x;
    if (i >= NPAD) return;
    idx[i] = i;
    if (i >= NCAND) key[i] = -3.0e38f;
}

__device__ __forceinline__ bool sort_before(float ka, int ia, float kb, int ib)
{   // true if a should come before b (descending score, ascending idx on tie)
    return (ka > kb) || (ka == kb && ia < ib);
}

__global__ void __launch_bounds__(256)
k_bitonic(float* __restrict__ key, int* __restrict__ idx, int j, int k)
{
    int i = blockIdx.x * blockDim.x + threadIdx.x;
    int ixj = i ^ j;
    if (ixj <= i || i >= NPAD) return;
    float ki = key[i], kj = key[ixj];
    int   ii = idx[i], ij = idx[ixj];
    bool dirDesc = ((i & k) == 0);
    bool jFirst = sort_before(kj, ij, ki, ii);
    bool doSwap = dirDesc ? jFirst : !jFirst;
    if (doSwap) { key[i] = kj; key[ixj] = ki; idx[i] = ij; idx[ixj] = ii; }
}

// ---------------------------------------------------------------------------
// 6) gather top-2000 + class-offset boxes for batched NMS
// ---------------------------------------------------------------------------
__global__ void __launch_bounds__(256)
k_gather(const float* __restrict__ key, const int* __restrict__ idx,
         const float* __restrict__ cbox, const int* __restrict__ ih_p,
         const int* __restrict__ iw_p, float* __restrict__ tscore,
         float* __restrict__ tbox, float* __restrict__ ob, int* __restrict__ tlabel)
{
    int i = blockIdx.x * blockDim.x + threadIdx.x;
    if (i >= PRENMS) return;
    float s = key[i];
    int id = idx[i];
    int label = (id % NFG) + 1;
    float off = (float)label * (fmaxf((float)ih_p[0], (float)iw_p[0]) + 1.f);
    tscore[i] = s;
    tlabel[i] = label;
    for (int d = 0; d < 4; ++d) {
        float b = cbox[(size_t)id * 4 + d];
        tbox[i * 4 + d] = b;
        ob[i * 4 + d] = b + off;
    }
}

// ---------------------------------------------------------------------------
// 7) sequential NMS over sorted top-2000 (single workgroup, boxes in LDS)
// ---------------------------------------------------------------------------
__global__ void __launch_bounds__(1024)
k_nms(const float* __restrict__ tscore, const float* __restrict__ ob,
      int* __restrict__ keep_g)
{
    __shared__ float bx[PRENMS * 4];
    __shared__ int   kp[PRENMS];
    int tid = threadIdx.x;
    for (int i = tid; i < PRENMS; i += 1024) {
        bx[i * 4 + 0] = ob[i * 4 + 0];
        bx[i * 4 + 1] = ob[i * 4 + 1];
        bx[i * 4 + 2] = ob[i * 4 + 2];
        bx[i * 4 + 3] = ob[i * 4 + 3];
        kp[i] = (tscore[i] > -0.5f) ? 1 : 0;
    }
    __syncthreads();
    for (int i = 0; i < PRENMS; ++i) {
        if (kp[i]) {  // uniform read
            float ax1 = bx[i * 4 + 0], ay1 = bx[i * 4 + 1];
            float ax2 = bx[i * 4 + 2], ay2 = bx[i * 4 + 3];
            float aarea = (ax2 - ax1) * (ay2 - ay1);
            for (int jj = i + 1 + tid; jj < PRENMS; jj += 1024) {
                if (kp[jj]) {
                    float x1 = fmaxf(ax1, bx[jj * 4 + 0]);
                    float y1 = fmaxf(ay1, bx[jj * 4 + 1]);
                    float x2 = fminf(ax2, bx[jj * 4 + 2]);
                    float y2 = fminf(ay2, bx[jj * 4 + 3]);
                    float iw = fmaxf(x2 - x1, 0.f), ihh = fmaxf(y2 - y1, 0.f);
                    float inter = iw * ihh;
                    float barea = (bx[jj * 4 + 2] - bx[jj * 4 + 0]) *
                                  (bx[jj * 4 + 3] - bx[jj * 4 + 1]);
                    float iou = inter / (aarea + barea - inter + 1e-9f);
                    if (iou > 0.5f) kp[jj] = 0;
                }
            }
        }
        __syncthreads();
    }
    for (int i = tid; i < PRENMS; i += 1024) keep_g[i] = kp[i];
}

// ---------------------------------------------------------------------------
// 8) final top-100 selection (matches lax.top_k tie semantics) + output write
//    out layout: boxes[100*4] | scores[100] | labels[100]  (600 floats)
// ---------------------------------------------------------------------------
__global__ void k_final(const int* __restrict__ keep, const float* __restrict__ tscore,
                        const float* __restrict__ tbox, const int* __restrict__ tlabel,
                        float* __restrict__ out)
{
    if (threadIdx.x != 0 || blockIdx.x != 0) return;
    int sel[TOPK];
    int cnt = 0;
    for (int i = 0; i < PRENMS && cnt < TOPK; ++i) if (keep[i]) sel[cnt++] = i;
    for (int i = 0; i < PRENMS && cnt < TOPK; ++i) if (!keep[i]) sel[cnt++] = i;
    for (int t = 0; t < TOPK; ++t) {
        int i = sel[t];
        out[t * 4 + 0] = tbox[i * 4 + 0];
        out[t * 4 + 1] = tbox[i * 4 + 1];
        out[t * 4 + 2] = tbox[i * 4 + 2];
        out[t * 4 + 3] = tbox[i * 4 + 3];
        out[400 + t] = keep[i] ? tscore[i] : 0.f;
        out[500 + t] = (float)tlabel[i];
    }
}

// ---------------------------------------------------------------------------
// Host: orchestrate on `stream` (graph-capture safe, deterministic)
// ---------------------------------------------------------------------------
static inline size_t align256(size_t x) { return (x + 255) & ~(size_t)255; }

extern "C" void kernel_launch(void* const* d_in, const int* in_sizes, int n_in,
                              void* d_out, int out_size, void* d_ws, size_t ws_size,
                              hipStream_t stream)
{
    const float* feat   = (const float*)d_in[0];
    const float* rois   = (const float*)d_in[1];
    const float* fc1_w  = (const float*)d_in[2];
    const float* fc1_b  = (const float*)d_in[3];
    const float* fc2_w  = (const float*)d_in[4];
    const float* fc2_b  = (const float*)d_in[5];
    const float* cls_w  = (const float*)d_in[6];
    const float* cls_b  = (const float*)d_in[7];
    const float* reg_w  = (const float*)d_in[8];
    const float* reg_b  = (const float*)d_in[9];
    const int*   im_h   = (const int*)d_in[10];
    const int*   im_w   = (const int*)d_in[11];
    float* out = (float*)d_out;

    // workspace carve-out (~94 MB total)
    char* ws = (char*)d_ws;
    size_t off = 0;
    __bf16* wt1    = (__bf16*)(ws + off); off = align256(off + (size_t)DDIM * FDIM * 2);
    __bf16* wt2    = (__bf16*)(ws + off); off = align256(off + (size_t)FDIM * FDIM * 2);
    __bf16* pooled = (__bf16*)(ws + off); off = align256(off + (size_t)MPAD * DDIM * 2);
    __bf16* x1     = (__bf16*)(ws + off); off = align256(off + (size_t)MPAD * FDIM * 2);
    float*  x2     = (float*)(ws + off);  off = align256(off + (size_t)MPAD * FDIM * 4);
    float*  heads  = (float*)(ws + off);  off = align256(off + (size_t)NROI * 105 * 4);
    float*  cscore = (float*)(ws + off);  off = align256(off + (size_t)NPAD * 4);
    int*    cidx   = (int*)(ws + off);    off = align256(off + (size_t)NPAD * 4);
    float*  cbox   = (float*)(ws + off);  off = align256(off + (size_t)NCAND * 4 * 4);
    float*  tscore = (float*)(ws + off);  off = align256(off + (size_t)PRENMS * 4);
    float*  tbox   = (float*)(ws + off);  off = align256(off + (size_t)PRENMS * 4 * 4);
    float*  obox   = (float*)(ws + off);  off = align256(off + (size_t)PRENMS * 4 * 4);
    int*    tlabel = (int*)(ws + off);    off = align256(off + (size_t)PRENMS * 4);
    int*    keep   = (int*)(ws + off);    off = align256(off + (size_t)PRENMS * 4);
    (void)ws_size; (void)n_in; (void)in_sizes; (void)out_size;

    // 0) weight conversion into fragment-packed bf16 layout + pad zeroing
    {
        long t1 = (long)FDIM * DDIM;
        k_pack_b<<<dim3((unsigned)((t1 + 255) / 256)), 256, 0, stream>>>(fc1_w, wt1, FDIM, DDIM);
        long t2 = (long)FDIM * FDIM;
        k_pack_b<<<dim3((unsigned)((t2 + 255) / 256)), 256, 0, stream>>>(fc2_w, wt2, FDIM, FDIM);
        long tz = (long)(MPAD - NROI) * DDIM;
        k_zero_pad<<<dim3((unsigned)((tz + 255) / 256)), 256, 0, stream>>>(pooled);
    }
    // 1) roi pool
    k_roi_pool<<<NROI, 256, 0, stream>>>(feat, rois, im_w, pooled);
    // 2) FC1 (WMMA bf16) -> bf16, FC2 (WMMA bf16) -> f32
    k_gemm_bf16<true><<<dim3(FDIM / BN, MPAD / BM), 256, 0, stream>>>(
        pooled, wt1, fc1_b, (void*)x1, MPAD, FDIM, DDIM);
    k_gemm_bf16<false><<<dim3(FDIM / BN, MPAD / BM), 256, 0, stream>>>(
        x1, wt2, fc2_b, (void*)x2, MPAD, FDIM, FDIM);
    // 3) heads
    k_heads<<<NROI, 128, 0, stream>>>(x2, cls_w, cls_b, reg_w, reg_b, heads);
    // 4) decode + filter, 5) sort
    k_sort_init<<<NPAD / 256, 256, 0, stream>>>(cscore, cidx);
    k_decode<<<(NROI + 255) / 256, 256, 0, stream>>>(heads, rois, im_h, im_w, cscore, cbox);
    for (int k = 2; k <= NPAD; k <<= 1)
        for (int j = k >> 1; j > 0; j >>= 1)
            k_bitonic<<<NPAD / 256, 256, 0, stream>>>(cscore, cidx, j, k);
    // 6) gather top-2000, 7) NMS, 8) final
    k_gather<<<(PRENMS + 255) / 256, 256, 0, stream>>>(cscore, cidx, cbox, im_h, im_w,
                                                       tscore, tbox, obox, tlabel);
    k_nms<<<1, 1024, 0, stream>>>(tscore, obox, keep);
    k_final<<<1, 64, 0, stream>>>(keep, tscore, tbox, tlabel, out);
}
